// EncodeProcessDecode_12043088297989
// MI455X (gfx1250) — compile-verified, matrix operand order checked by reading, not run on hardware
//
#include <hip/hip_runtime.h>
#include <stdint.h>

typedef __attribute__((ext_vector_type(16))) __bf16 v16bf;
typedef __attribute__((ext_vector_type(8)))  float  v8f;

#define NNODES 16000
#define NEDGES 100000

enum { EPI_RELU = 0, EPI_LN_STORE = 1, EPI_LN_EDGE = 2, EPI_LN_NODE = 3 };

__device__ __forceinline__ unsigned short f2bf(float x) {
  unsigned u = __float_as_uint(x);
  unsigned r = ((u >> 16) & 1u) + 0x7fffu;  // round-to-nearest-even
  return (unsigned short)((u + r) >> 16);
}
__device__ __forceinline__ float bf2f(unsigned short h) {
  return __uint_as_float(((unsigned)h) << 16);
}
// CDNA5 16-bit A/B fragment K index for element e (0..15), half hi (lane>>4)
__device__ __forceinline__ int kmap(int e, int hi) {
  int vg = e >> 1;
  return ((vg < 4) ? 0 : 16) + hi * 8 + (vg & 3) * 2 + (e & 1);
}

// Pack f32 weight [Kreal x 128] (row-major) into WMMA bf16 B-fragment order:
// packed[((kt*8+ct)*32+lane)*16 + e]. Zero-pads K up to gridDim.x*32.
__global__ void pack_w_kernel(const float* __restrict__ W,
                              unsigned short* __restrict__ out, int Kreal) {
  int t = blockIdx.x * 256 + threadIdx.x;  // grid.x == Ktiles
  int lane = t & 31, ct = (t >> 5) & 7, kt = t >> 8;
  int hi = lane >> 4, n = ct * 16 + (lane & 15);
  unsigned short* o = out + (size_t)t * 16;
#pragma unroll
  for (int e = 0; e < 16; ++e) {
    int k = kt * 32 + kmap(e, hi);
    o[e] = (k < Kreal) ? f2bf(W[(size_t)k * 128 + n]) : (unsigned short)0;
  }
}

// Pad raw features [rows x w] f32 -> [rows x 32] bf16 (zero-padded)
__global__ void pad_feat_kernel(const float* __restrict__ X,
                                unsigned short* __restrict__ Y, int rows, int w) {
  int t = blockIdx.x * 256 + threadIdx.x;
  if (t >= rows * 32) return;
  int r = t >> 5, k = t & 31;
  Y[t] = (k < w) ? f2bf(X[(size_t)r * w + k]) : (unsigned short)0;
}

__global__ void zero_f32_kernel(float* __restrict__ p, int n) {
  int t = blockIdx.x * 256 + threadIdx.x;
  if (t < n) p[t] = 0.f;
}

__global__ void f32_to_bf16_kernel(const float* __restrict__ x,
                                   unsigned short* __restrict__ y, int n) {
  int t = blockIdx.x * 256 + threadIdx.x;
  if (t < n) y[t] = f2bf(x[t]);
}

// Fused GEMM: Y[rows x 128] = epilogue( concat(A0[idx0],A1[idx1],A2) @ B + bias )
// One block = 16 rows x 128 cols, 8 waves, each wave one 16x16 WMMA tile.
// W0/W1/W2 (concat segment widths, multiples of 32) and MODE are compile-time,
// so the K loop fully unrolls with no per-tile branching.
template <int W0, int W1, int W2, int MODE>
__global__ __launch_bounds__(256) void gemm_mlp(
    const unsigned short* __restrict__ A0, const int* __restrict__ idx0,
    const unsigned short* __restrict__ A1, const int* __restrict__ idx1,
    const unsigned short* __restrict__ A2,
    const unsigned short* __restrict__ Bp,
    const float* __restrict__ bias,
    const float* __restrict__ gamma, const float* __restrict__ beta,
    unsigned short* __restrict__ outB,  // bf16 out [rows x 128]
    float* __restrict__ outF,           // f32 master (LN modes)
    float* __restrict__ aggF,           // segment-sum target (edge mode)
    const int* __restrict__ recv) {
  constexpr int KT = (W0 + W1 + W2) / 32;
  __shared__ float tile[16 * 128];
  const int t = threadIdx.x;
  const int lane = t & 31;
  const int wave = t >> 5;  // column tile 0..7
  const int m = lane & 15;
  const int hi = lane >> 4;
  const size_t rowg = (size_t)blockIdx.x * 16 + m;

  // Per-lane gathered row base pointers for each concat segment.
  const unsigned short* base0 = A0 + (size_t)(idx0 ? idx0[rowg] : (int)rowg) * W0;
  const unsigned short* base1 = nullptr;
  const unsigned short* base2 = nullptr;
  if constexpr (W1 > 0) base1 = A1 + (size_t)(idx1 ? idx1[rowg] : (int)rowg) * W1;
  if constexpr (W2 > 0) base2 = A2 + rowg * W2;

  v8f acc = {};
#pragma unroll
  for (int kt = 0; kt < KT; ++kt) {
    const int kbase = kt * 32;  // segment choice folds at compile time
    const unsigned short* p;
    if (kbase < W0)           p = base0 + kbase;
    else if (kbase < W0 + W1) p = base1 + (kbase - W0);
    else                      p = base2 + (kbase - W0 - W1);
    p += hi * 8;
    union { unsigned u[8]; v16bf v; } af;
#pragma unroll
    for (int j = 0; j < 4; ++j) af.u[j]     = *(const unsigned*)(p + 2 * j);
#pragma unroll
    for (int j = 0; j < 4; ++j) af.u[4 + j] = *(const unsigned*)(p + 16 + 2 * j);
    const v16bf bfrag = *(const v16bf*)(Bp + ((size_t)(kt * 8 + wave) * 32 + lane) * 16);
    acc = __builtin_amdgcn_wmma_f32_16x16x32_bf16(false, af.v, false, bfrag,
                                                  (short)0, acc, false, false);
  }

  // C layout: lane (0-15|16-31) = col n, VGPR v = row v (+8 for hi half)
#pragma unroll
  for (int v = 0; v < 8; ++v)
    tile[(v + 8 * hi) * 128 + wave * 16 + m] = acc[v];
  __syncthreads();

  const int row = t >> 4;  // 0..15
  const int c0 = t & 15;   // this thread covers cols c0, c0+16, ..., c0+112
  const size_t rg = (size_t)blockIdx.x * 16 + row;

  if constexpr (MODE == EPI_RELU) {
#pragma unroll
    for (int j = 0; j < 8; ++j) {
      int n = c0 + j * 16;
      float y = tile[row * 128 + n] + bias[n];
      outB[rg * 128 + n] = f2bf(fmaxf(y, 0.f));
    }
  } else {
    float vals[8], s = 0.f, s2 = 0.f;
#pragma unroll
    for (int j = 0; j < 8; ++j) {
      int n = c0 + j * 16;
      float x = tile[row * 128 + n] + bias[n];
      vals[j] = x; s += x; s2 += x * x;
    }
    // 16-lane row reduction (xor masks 1..8 stay inside each wave32 half)
#pragma unroll
    for (int off = 1; off < 16; off <<= 1) {
      s  += __shfl_xor(s, off, 32);
      s2 += __shfl_xor(s2, off, 32);
    }
    const float mu = s * (1.f / 128.f);
    const float rs = rsqrtf(s2 * (1.f / 128.f) - mu * mu + 1e-5f);
    int rv = 0;
    if constexpr (MODE == EPI_LN_EDGE) rv = recv[rg];
#pragma unroll
    for (int j = 0; j < 8; ++j) {
      int n = c0 + j * 16;
      float ln = (vals[j] - mu) * rs * gamma[n] + beta[n];
      size_t o = rg * 128 + n;
      if constexpr (MODE == EPI_LN_STORE) {        // encoder: store
        outF[o] = ln; outB[o] = f2bf(ln);
      } else if constexpr (MODE == EPI_LN_NODE) {  // nodes += new_nodes
        float nf = outF[o] + ln; outF[o] = nf; outB[o] = f2bf(nf);
      } else {                                     // edges += new; scatter to agg
        float ef = outF[o] + ln; outF[o] = ef; outB[o] = f2bf(ef);
        atomicAdd(&aggF[(size_t)rv * 128 + n], ln);
      }
    }
  }
}

// Final decoder layer: [rows x 128] bf16 @ [128 x 3] f32 + b -> f32 out
__global__ void dec_final_kernel(const unsigned short* __restrict__ H,
                                 const float* __restrict__ W,
                                 const float* __restrict__ b,
                                 float* __restrict__ out, int rows) {
  int t = blockIdx.x * 256 + threadIdx.x;
  if (t >= rows * 3) return;
  int r = t / 3, c = t - r * 3;
  float s = b[c];
#pragma unroll 8
  for (int k = 0; k < 128; ++k) s += bf2f(H[(size_t)r * 128 + k]) * W[k * 3 + c];
  out[t] = s;
}

extern "C" void kernel_launch(void* const* d_in, const int* in_sizes, int n_in,
                              void* d_out, int out_size, void* d_ws, size_t ws_size,
                              hipStream_t stream) {
  (void)in_sizes; (void)n_in; (void)out_size; (void)ws_size;
  const float* nodeF = (const float*)d_in[0];
  const float* edgeF = (const float*)d_in[1];
  const int* senders = (const int*)d_in[2];
  const int* receivers = (const int*)d_in[3];
  // params leaf order: blocks[10]{edge{W0,b0,W1,b1,W2,b2,g,be}, node{...}}, dec{W0..b2}, enc_edge{8}, enc_node{8}
  const int P0 = 4;
  auto blkP = [&](int b, int node_mlp, int leaf) { return P0 + b * 16 + node_mlp * 8 + leaf; };
  const int DEC0 = P0 + 160, ENCE0 = DEC0 + 6, ENCN0 = ENCE0 + 8;
  auto F = [&](int i) { return (const float*)d_in[i]; };

  char* wsp = (char*)d_ws;
  auto alloc = [&](size_t bytes) {
    void* p = (void*)wsp; wsp += (bytes + 255) & ~(size_t)255; return p;
  };
  unsigned short* nodeFp = (unsigned short*)alloc((size_t)NNODES * 32 * 2);
  unsigned short* edgeFp = (unsigned short*)alloc((size_t)NEDGES * 32 * 2);
  float*          nodesF = (float*)alloc((size_t)NNODES * 128 * 4);
  unsigned short* nodesB = (unsigned short*)alloc((size_t)NNODES * 128 * 2);
  float*          edgesF = (float*)alloc((size_t)NEDGES * 128 * 4);
  unsigned short* edgesB = (unsigned short*)alloc((size_t)NEDGES * 128 * 2);
  unsigned short* hidA   = (unsigned short*)alloc((size_t)NEDGES * 128 * 2);
  unsigned short* hidB   = (unsigned short*)alloc((size_t)NEDGES * 128 * 2);
  float*          aggF   = (float*)alloc((size_t)NNODES * 128 * 4);
  unsigned short* aggB   = (unsigned short*)alloc((size_t)NNODES * 128 * 2);

  auto packW = [&](int pidx, int Kreal, int Kpad) -> unsigned short* {
    int Ktiles = Kpad / 32;
    unsigned short* dst = (unsigned short*)alloc((size_t)Kpad * 128 * 2);
    pack_w_kernel<<<Ktiles, 256, 0, stream>>>(F(pidx), dst, Kreal);
    return dst;
  };
  unsigned short *pk_en0 = packW(ENCN0 + 0, 12, 32),
                 *pk_en1 = packW(ENCN0 + 2, 128, 128),
                 *pk_en2 = packW(ENCN0 + 4, 128, 128);
  unsigned short *pk_ee0 = packW(ENCE0 + 0, 7, 32),
                 *pk_ee1 = packW(ENCE0 + 2, 128, 128),
                 *pk_ee2 = packW(ENCE0 + 4, 128, 128);
  unsigned short *pk_e0[10], *pk_e1[10], *pk_e2[10], *pk_n0[10], *pk_n1[10], *pk_n2[10];
  for (int b = 0; b < 10; ++b) {
    pk_e0[b] = packW(blkP(b, 0, 0), 384, 384);
    pk_e1[b] = packW(blkP(b, 0, 2), 128, 128);
    pk_e2[b] = packW(blkP(b, 0, 4), 128, 128);
    pk_n0[b] = packW(blkP(b, 1, 0), 256, 256);
    pk_n1[b] = packW(blkP(b, 1, 2), 128, 128);
    pk_n2[b] = packW(blkP(b, 1, 4), 128, 128);
  }
  unsigned short *pk_d0 = packW(DEC0 + 0, 128, 128), *pk_d1 = packW(DEC0 + 2, 128, 128);

  pad_feat_kernel<<<(NNODES * 32 + 255) / 256, 256, 0, stream>>>(nodeF, nodeFp, NNODES, 12);
  pad_feat_kernel<<<(NEDGES * 32 + 255) / 256, 256, 0, stream>>>(edgeF, edgeFp, NEDGES, 7);

  // ---- Encoders ----
  gemm_mlp<32, 0, 0, EPI_RELU><<<NNODES / 16, 256, 0, stream>>>(
      nodeFp, nullptr, nullptr, nullptr, nullptr, pk_en0,
      F(ENCN0 + 1), nullptr, nullptr, hidA, nullptr, nullptr, nullptr);
  gemm_mlp<128, 0, 0, EPI_RELU><<<NNODES / 16, 256, 0, stream>>>(
      hidA, nullptr, nullptr, nullptr, nullptr, pk_en1,
      F(ENCN0 + 3), nullptr, nullptr, hidB, nullptr, nullptr, nullptr);
  gemm_mlp<128, 0, 0, EPI_LN_STORE><<<NNODES / 16, 256, 0, stream>>>(
      hidB, nullptr, nullptr, nullptr, nullptr, pk_en2,
      F(ENCN0 + 5), F(ENCN0 + 6), F(ENCN0 + 7), nodesB, nodesF, nullptr, nullptr);

  gemm_mlp<32, 0, 0, EPI_RELU><<<NEDGES / 16, 256, 0, stream>>>(
      edgeFp, nullptr, nullptr, nullptr, nullptr, pk_ee0,
      F(ENCE0 + 1), nullptr, nullptr, hidA, nullptr, nullptr, nullptr);
  gemm_mlp<128, 0, 0, EPI_RELU><<<NEDGES / 16, 256, 0, stream>>>(
      hidA, nullptr, nullptr, nullptr, nullptr, pk_ee1,
      F(ENCE0 + 3), nullptr, nullptr, hidB, nullptr, nullptr, nullptr);
  gemm_mlp<128, 0, 0, EPI_LN_STORE><<<NEDGES / 16, 256, 0, stream>>>(
      hidB, nullptr, nullptr, nullptr, nullptr, pk_ee2,
      F(ENCE0 + 5), F(ENCE0 + 6), F(ENCE0 + 7), edgesB, edgesF, nullptr, nullptr);

  // ---- Processor: 10 message-passing steps ----
  for (int b = 0; b < 10; ++b) {
    zero_f32_kernel<<<(NNODES * 128 + 255) / 256, 256, 0, stream>>>(aggF, NNODES * 128);
    // edge MLP on concat(nodes[s], nodes[r], edges), K = 384
    gemm_mlp<128, 128, 128, EPI_RELU><<<NEDGES / 16, 256, 0, stream>>>(
        nodesB, senders, nodesB, receivers, edgesB, pk_e0[b],
        F(blkP(b, 0, 1)), nullptr, nullptr, hidA, nullptr, nullptr, nullptr);
    gemm_mlp<128, 0, 0, EPI_RELU><<<NEDGES / 16, 256, 0, stream>>>(
        hidA, nullptr, nullptr, nullptr, nullptr, pk_e1[b],
        F(blkP(b, 0, 3)), nullptr, nullptr, hidB, nullptr, nullptr, nullptr);
    gemm_mlp<128, 0, 0, EPI_LN_EDGE><<<NEDGES / 16, 256, 0, stream>>>(
        hidB, nullptr, nullptr, nullptr, nullptr, pk_e2[b],
        F(blkP(b, 0, 5)), F(blkP(b, 0, 6)), F(blkP(b, 0, 7)),
        edgesB, edgesF, aggF, receivers);
    f32_to_bf16_kernel<<<(NNODES * 128 + 255) / 256, 256, 0, stream>>>(aggF, aggB, NNODES * 128);
    // node MLP on concat(nodes, agg), K = 256
    gemm_mlp<128, 128, 0, EPI_RELU><<<NNODES / 16, 256, 0, stream>>>(
        nodesB, nullptr, aggB, nullptr, nullptr, pk_n0[b],
        F(blkP(b, 1, 1)), nullptr, nullptr, hidA, nullptr, nullptr, nullptr);
    gemm_mlp<128, 0, 0, EPI_RELU><<<NNODES / 16, 256, 0, stream>>>(
        hidA, nullptr, nullptr, nullptr, nullptr, pk_n1[b],
        F(blkP(b, 1, 3)), nullptr, nullptr, hidB, nullptr, nullptr, nullptr);
    gemm_mlp<128, 0, 0, EPI_LN_NODE><<<NNODES / 16, 256, 0, stream>>>(
        hidB, nullptr, nullptr, nullptr, nullptr, pk_n2[b],
        F(blkP(b, 1, 5)), F(blkP(b, 1, 6)), F(blkP(b, 1, 7)),
        nodesB, nodesF, nullptr, nullptr);
  }

  // ---- Decoder ----
  gemm_mlp<128, 0, 0, EPI_RELU><<<NNODES / 16, 256, 0, stream>>>(
      nodesB, nullptr, nullptr, nullptr, nullptr, pk_d0,
      F(DEC0 + 1), nullptr, nullptr, hidA, nullptr, nullptr, nullptr);
  gemm_mlp<128, 0, 0, EPI_RELU><<<NNODES / 16, 256, 0, stream>>>(
      hidA, nullptr, nullptr, nullptr, nullptr, pk_d1,
      F(DEC0 + 3), nullptr, nullptr, hidB, nullptr, nullptr, nullptr);
  dec_final_kernel<<<(NNODES * 3 + 255) / 256, 256, 0, stream>>>(
      hidB, F(DEC0 + 4), F(DEC0 + 5), (float*)d_out, NNODES);
}
